// NoFoDifformer_mk_7670811591188
// MI455X (gfx1250) — matrix operand order repaired and checked by reading, not compiled, not run on hardware
//
#include <hip/hip_runtime.h>

// ---------------------------------------------------------------------------
// NoFoDifformer forward on MI455X (gfx1250, wave32, WMMA).
// Roofline: dominated by streaming the 400 MB f32 `u` matrix twice (u^T@h and
// u@(e.*utx)) -> 800 MB -> ~34 us floor @ 23.3 TB/s.  All GEMMs use one
// templated WMMA kernel with a register->LDS double-buffered pipeline:
//   global_load_b128 (next tile) | ds_load_b128 + v_wmma (current tile)
//   | cvt f32->f16 + ds_store (next tile) | one s_barrier per tile.
// LDS tiles are stored directly in WMMA fragment order (ISA 7.12.2) so each
// lane's fragment read is two ds_load_b128.
// ---------------------------------------------------------------------------

#define NN      10000
#define CC      64
#define HIDN    128
#define NFEATN  512
#define KPOW    10
#define NFRQ    16

typedef __attribute__((ext_vector_type(16))) _Float16 v16h;
typedef __attribute__((ext_vector_type(4)))  _Float16 h4;
typedef __attribute__((ext_vector_type(8)))  float    v8f;
typedef __attribute__((ext_vector_type(4)))  float    f4;

struct Ptrs {
    const float* a;
    const float* b0; const float* b1; const float* b2; const float* b3;
    const float* rs;
};

// Per-thread tile loads (one f4 of A + four f4 of B), pointers pre-advanced.
template<bool AT, bool RS, bool GUARD>
__device__ __forceinline__ void load_tile(const Ptrs& p, int rem, int tid,
                                          f4& ra, f4 (&rb)[4])
{
    if (!AT) {
        int k4 = (tid & 7) << 2;
        ra = f4{0.0f, 0.0f, 0.0f, 0.0f};
        if (!GUARD || (k4 + 3) < rem) ra = *(const f4*)p.a;
        else {
            #pragma unroll
            for (int j = 0; j < 4; ++j)
                if ((k4 + j) < rem) ra[j] = p.a[j];
        }
    } else {
        int k = tid >> 2;
        ra = f4{0.0f, 0.0f, 0.0f, 0.0f};
        if (!GUARD || k < rem) ra = *(const f4*)p.a;
    }
    const float* bp[4] = {p.b0, p.b1, p.b2, p.b3};
    #pragma unroll
    for (int i = 0; i < 4; ++i) {
        int k = i * 8 + (tid >> 4);
        rb[i] = f4{0.0f, 0.0f, 0.0f, 0.0f};
        if (!GUARD || k < rem) {
            rb[i] = *(const f4*)bp[i];
            if (RS) {
                float sc = p.rs[i * 8];
                rb[i][0] *= sc; rb[i][1] *= sc; rb[i][2] *= sc; rb[i][3] *= sc;
            }
        }
    }
}

template<bool AT, bool RS>
__device__ __forceinline__ void advance(Ptrs& p, int lda, int ldb)
{
    p.a += AT ? (size_t)32 * lda : (size_t)32;
    size_t bs = (size_t)32 * ldb;
    p.b0 += bs; p.b1 += bs; p.b2 += bs; p.b3 += bs;
    if (RS) p.rs += 32;
}

// Convert to f16 and scatter into WMMA-fragment-ordered LDS (ISA 7.12.2):
//   A (16x32): (m,k) -> lane = m + 16*((k>>3)&1), e = 8*(k>>4) + (k&7)
//   B (32x16): (k,n) -> lane = n + 16*((k>>3)&1), e = 8*(k>>4) + (k&7)
template<bool AT>
__device__ __forceinline__ void store_tile(const f4& ra, const f4 (&rb)[4],
    _Float16* __restrict__ Afrag, _Float16* __restrict__ Bfrag, int tid)
{
    if (!AT) {
        int m  = tid >> 3;
        int k4 = (tid & 7) << 2;
        int lane  = m + (((k4 >> 3) & 1) << 4);
        int ebase = ((k4 >> 4) << 3) + (k4 & 7);
        h4 hv;
        #pragma unroll
        for (int j = 0; j < 4; ++j) hv[j] = (_Float16)ra[j];
        *(h4*)&Afrag[lane * 16 + ebase] = hv;            // ds_store_b64
    } else {
        int k  = tid >> 2;
        int m4 = (tid & 3) << 2;
        int hi = ((k >> 3) & 1) << 4;
        int e  = ((k >> 4) << 3) + (k & 7);
        #pragma unroll
        for (int j = 0; j < 4; ++j)
            Afrag[(m4 + j + hi) * 16 + e] = (_Float16)ra[j];
    }
    int c4 = (tid & 15) << 2;
    int w  = c4 >> 4;
    int nb = c4 & 15;
    #pragma unroll
    for (int i = 0; i < 4; ++i) {
        int k  = i * 8 + (tid >> 4);
        int hi = ((k >> 3) & 1) << 4;
        int e  = ((k >> 4) << 3) + (k & 7);
        #pragma unroll
        for (int j = 0; j < 4; ++j)
            Bfrag[w * 512 + (nb + j + hi) * 16 + e] = (_Float16)rb[i][j];
    }
}

// ---------------------------------------------------------------------------
// Generic WMMA GEMM: out[M,Nc] = epilogue( op(A)[M,K] @ (rowscale.*B)[K,Nc] )
// block = 128 threads (4 waves), block tile 16x64 (one 16x16 tile / wave),
// K step 32, double-buffered pipeline, single guarded tail tile.
// atomic_out: split-K partial accumulation via global_atomic_add_f32.
// act: 0 none, 1 relu, 2 exact gelu.
// ---------------------------------------------------------------------------
template<bool AT, bool RS>
__global__ __launch_bounds__(128) void gemm_wmma(
    const float* __restrict__ A, int lda,
    const float* __restrict__ B, int ldb,
    const float* __restrict__ rowscale,
    const float* __restrict__ bias,
    const float* __restrict__ resid,
    float* __restrict__ out, int ldo,
    int M, int Nc, int Kdim,
    int act, int kchunk, int atomic_out)
{
    alignas(32) __shared__ _Float16 AfragBuf[2][32 * 16];       // 2 x 1 KB
    alignas(32) __shared__ _Float16 BfragBuf[2][4 * 32 * 16];   // 2 x 4 KB

    const int tid  = threadIdx.x;
    const int lane = tid & 31;
    const int wave = tid >> 5;
    const int m0   = blockIdx.x * 16;
    const int n0   = blockIdx.y * 64;

    int kbeg = blockIdx.z * kchunk;
    int kend = kbeg + kchunk;
    if (kend > Kdim) kend = Kdim;
    const int klen  = kend - kbeg;
    const int nfull = klen >> 5;
    const int rem   = klen & 31;

    // per-thread streaming pointers (constant byte stride per k-tile)
    Ptrs p;
    if (!AT) p.a = A + (size_t)(m0 + (tid >> 3)) * lda + kbeg + ((tid & 7) << 2);
    else     p.a = A + (size_t)(kbeg + (tid >> 2)) * lda + m0 + ((tid & 3) << 2);
    {
        int c4 = (tid & 15) << 2;
        int kb = kbeg + (tid >> 4);
        p.b0 = B + (size_t)(kb +  0) * ldb + n0 + c4;
        p.b1 = B + (size_t)(kb +  8) * ldb + n0 + c4;
        p.b2 = B + (size_t)(kb + 16) * ldb + n0 + c4;
        p.b3 = B + (size_t)(kb + 24) * ldb + n0 + c4;
    }
    p.rs = RS ? (rowscale + kbeg + (tid >> 4)) : nullptr;

    v8f acc = {};
    f4 ra; f4 rb[4];
    int cur = 0;

    if (nfull > 0) {
        load_tile<AT, RS, false>(p, 32, tid, ra, rb);
        advance<AT, RS>(p, lda, ldb);
        store_tile<AT>(ra, rb, AfragBuf[0], BfragBuf[0], tid);
        for (int it = 0; it < nfull; ++it) {
            __syncthreads();                         // buf[cur] visible to all
            const bool more = (it + 1) < nfull;
            if (more) {
                __builtin_prefetch(p.a + (AT ? (size_t)32 * lda : (size_t)32), 0, 1);
                load_tile<AT, RS, false>(p, 32, tid, ra, rb);   // next tile -> regs
                advance<AT, RS>(p, lda, ldb);
            }
            v16h af = *(const v16h*)&AfragBuf[cur][lane * 16];          // 2x b128
            v16h bf = *(const v16h*)&BfragBuf[cur][wave * 512 + lane * 16];
            acc = __builtin_amdgcn_wmma_f32_16x16x32_f16(
                      false, af, false, bf, (short)0, acc, false, false);
            if (more)
                store_tile<AT>(ra, rb, AfragBuf[cur ^ 1], BfragBuf[cur ^ 1], tid);
            cur ^= 1;
        }
    }
    if (rem) {                                       // K % 32 == 16 tail
        load_tile<AT, RS, true>(p, rem, tid, ra, rb);
        store_tile<AT>(ra, rb, AfragBuf[cur], BfragBuf[cur], tid);
        __syncthreads();
        v16h af = *(const v16h*)&AfragBuf[cur][lane * 16];
        v16h bf = *(const v16h*)&BfragBuf[cur][wave * 512 + lane * 16];
        acc = __builtin_amdgcn_wmma_f32_16x16x32_f16(
                  false, af, false, bf, (short)0, acc, false, false);
    }

    // epilogue (C/D layout: vgpr r -> row r + 8*(lane>=16), col = lane&15)
    const int n     = n0 + wave * 16 + (lane & 15);
    const int mbase = m0 + ((lane >> 4) << 3);
    if (atomic_out) {
        #pragma unroll
        for (int r = 0; r < 8; ++r)
            atomicAdd(&out[(size_t)(mbase + r) * ldo + n], acc[r]);
    } else {
        float bv = bias ? bias[n] : 0.0f;
        #pragma unroll
        for (int r = 0; r < 8; ++r) {
            int m = mbase + r;
            float v = acc[r] + bv;
            if (act == 1)      v = fmaxf(v, 0.0f);
            else if (act == 2) v = 0.5f * v * (1.0f + erff(v * 0.70710678118f));
            if (resid) v += resid[(size_t)m * ldo + n];
            out[(size_t)m * ldo + n] = v;
        }
    }
}

static void gemm_launch(bool at, bool rs, dim3 grid, hipStream_t st,
    const float* A, int lda, const float* B, int ldb,
    const float* rowscale, const float* bias, const float* resid,
    float* out, int ldo, int M, int Nc, int Kdim,
    int act, int kchunk, int atomic_out)
{
    dim3 blk(128);
    if (!at && !rs)
        gemm_wmma<false, false><<<grid, blk, 0, st>>>(A, lda, B, ldb, rowscale,
            bias, resid, out, ldo, M, Nc, Kdim, act, kchunk, atomic_out);
    else if (at && !rs)
        gemm_wmma<true, false><<<grid, blk, 0, st>>>(A, lda, B, ldb, rowscale,
            bias, resid, out, ldo, M, Nc, Kdim, act, kchunk, atomic_out);
    else if (!at && rs)
        gemm_wmma<false, true><<<grid, blk, 0, st>>>(A, lda, B, ldb, rowscale,
            bias, resid, out, ldo, M, Nc, Kdim, act, kchunk, atomic_out);
    else
        gemm_wmma<true, true><<<grid, blk, 0, st>>>(A, lda, B, ldb, rowscale,
            bias, resid, out, ldo, M, Nc, Kdim, act, kchunk, atomic_out);
}

// ---------------------------------------------------------------------------
// Row-wise LayerNorm over C=64, one wave per row (wave32), 8 rows / block.
// ---------------------------------------------------------------------------
__global__ __launch_bounds__(256) void ln_rows64(
    const float* __restrict__ in, const float* __restrict__ g,
    const float* __restrict__ b, float* __restrict__ outp, int M)
{
    int row  = blockIdx.x * 8 + (threadIdx.x >> 5);
    int lane = threadIdx.x & 31;
    if (row >= M) return;
    const float* x = in + (size_t)row * 64;
    float x0 = x[lane], x1 = x[lane + 32];
    float s = x0 + x1;
    for (int o = 16; o > 0; o >>= 1) s += __shfl_xor(s, o, 32);
    float mean = s * (1.0f / 64.0f);
    float d0 = x0 - mean, d1 = x1 - mean;
    float v = d0 * d0 + d1 * d1;
    for (int o = 16; o > 0; o >>= 1) v += __shfl_xor(v, o, 32);
    float rstd = rsqrtf(v * (1.0f / 64.0f) + 1e-5f);
    float* op = outp + (size_t)row * 64;
    op[lane]      = g[lane]      * d0 * rstd + b[lane];
    op[lane + 32] = g[lane + 32] * d1 * rstd + b[lane + 32];
}

// ---------------------------------------------------------------------------
// Differential-attention combine: xa = LN(xa1 - lam*xa2) * (1-lam_init)
// ---------------------------------------------------------------------------
__global__ __launch_bounds__(256) void attn_combine(
    const float* __restrict__ xa1, const float* __restrict__ xa2,
    const float* __restrict__ scal, const float* __restrict__ g,
    const float* __restrict__ b, float* __restrict__ outp, int M)
{
    int row  = blockIdx.x * 8 + (threadIdx.x >> 5);
    int lane = threadIdx.x & 31;
    if (row >= M) return;
    float lam = scal[0], post = scal[1];
    size_t o = (size_t)row * 64;
    float y0 = xa1[o + lane]      - lam * xa2[o + lane];
    float y1 = xa1[o + lane + 32] - lam * xa2[o + lane + 32];
    float s = y0 + y1;
    for (int off = 16; off > 0; off >>= 1) s += __shfl_xor(s, off, 32);
    float mean = s * (1.0f / 64.0f);
    float d0 = y0 - mean, d1 = y1 - mean;
    float v = d0 * d0 + d1 * d1;
    for (int off = 16; off > 0; off >>= 1) v += __shfl_xor(v, off, 32);
    float rstd = rsqrtf(v * (1.0f / 64.0f) + 1e-5f);
    outp[o + lane]      = (g[lane]      * d0 * rstd + b[lane])      * post;
    outp[o + lane + 32] = (g[lane + 32] * d1 * rstd + b[lane + 32]) * post;
}

// ---------------------------------------------------------------------------
// ParamNonHarmonicEncoding -> new_e[n]  (pure VALU, tiny)
// ---------------------------------------------------------------------------
__global__ __launch_bounds__(256) void eig_encode(
    const float* __restrict__ e, const float* __restrict__ fdel,
    const float* __restrict__ fbias, const float* __restrict__ rw,
    const float* __restrict__ rb, const float* __restrict__ aw,
    float* __restrict__ new_e, int M)
{
    int i = blockIdx.x * blockDim.x + threadIdx.x;
    if (i >= M) return;
    float fr[NFRQ];
    float c = 0.0f;
    #pragma unroll
    for (int f = 0; f < NFRQ; ++f) {
        float d = log1pf(expf(fdel[f])) + 0.25f;     // softplus + DMIN
        c += d;
        fr[f] = 50.0f * tanhf((c + fbias[0]) * (1.0f / 50.0f));
    }
    float ev = e[i];
    float p = 1.0f, acc = 0.0f;
    #pragma unroll
    for (int k = 0; k < KPOW; ++k) {
        p *= ev;                                     // e^(k+1)
        const float* rwk = rw + k * (1 + 2 * NFRQ);
        float s = rb[k] + rwk[0];
        #pragma unroll
        for (int f = 0; f < NFRQ; ++f) {
            float ph = p * fr[f];
            s += (sinf(ph) * rwk[1 + f] + cosf(ph) * rwk[1 + NFRQ + f]) * 0.25f;
        }
        acc += s * aw[k];
    }
    new_e[i] = acc;
}

// ---------------------------------------------------------------------------
// lambda scalars: scal[0] = exp(lq1.lk1) - exp(lq2.lk2) + lam_init
//                 scal[1] = 1 - lam_init           (lam_init = 0.2, layer 0)
// ---------------------------------------------------------------------------
__global__ __launch_bounds__(64) void lam_kernel(
    const float* __restrict__ lq1, const float* __restrict__ lk1,
    const float* __restrict__ lq2, const float* __restrict__ lk2,
    float* __restrict__ scal)
{
    __shared__ float red[2];
    __shared__ float red2[2];
    int t = threadIdx.x, lane = t & 31, w = t >> 5;
    float s1 = lq1[t] * lk1[t];
    float s2 = lq2[t] * lk2[t];
    for (int o = 16; o > 0; o >>= 1) { s1 += __shfl_xor(s1, o, 32); s2 += __shfl_xor(s2, o, 32); }
    if (lane == 0) { red[w] = s1; red2[w] = s2; }
    __syncthreads();
    if (t == 0) {
        float a = red[0] + red[1], b = red2[0] + red2[1];
        const float lam_init = 0.2f;                  // 0.8 - 0.6*exp(0)
        scal[0] = expf(a) - expf(b) + lam_init;
        scal[1] = 1.0f - lam_init;
    }
}

__device__ __forceinline__ float block64_sum(float v, float* red) {
    for (int o = 16; o > 0; o >>= 1) v += __shfl_xor(v, o, 32);
    int w = threadIdx.x >> 5;
    if ((threadIdx.x & 31) == 0) red[w] = v;
    __syncthreads();
    float r = red[0] + red[1];
    __syncthreads();
    return r;
}

// ---------------------------------------------------------------------------
// Gated mixing + FFN pre-norm, one 64-thread block per row.
// ---------------------------------------------------------------------------
__global__ __launch_bounds__(64) void gate_mix(
    const float* __restrict__ h, const float* __restrict__ mha,
    const float* __restrict__ hfur,
    const float* __restrict__ gg, const float* __restrict__ gb,
    const float* __restrict__ gw, const float* __restrict__ gbias,
    const float* __restrict__ fg, const float* __restrict__ fb,
    float* __restrict__ mix_out, float* __restrict__ lnmix_out, int M)
{
    __shared__ float red[2];
    int row = blockIdx.x, t = threadIdx.x;
    size_t o = (size_t)row * 64;
    float c0 = h[o + t], c1 = mha[o + t], c2 = hfur[o + t];

    float mean = block64_sum(c0 + c1 + c2, red) * (1.0f / 192.0f);
    float d0 = c0 - mean, d1 = c1 - mean, d2 = c2 - mean;
    float var = block64_sum(d0 * d0 + d1 * d1 + d2 * d2, red) * (1.0f / 192.0f);
    float rstd = rsqrtf(var + 1e-5f);
    float l0 = gg[t]       * d0 * rstd + gb[t];
    float l1 = gg[64 + t]  * d1 * rstd + gb[64 + t];
    float l2 = gg[128 + t] * d2 * rstd + gb[128 + t];

    float logit[3];
    #pragma unroll
    for (int j = 0; j < 3; ++j) {
        float part = l0 * gw[t * 3 + j] + l1 * gw[(64 + t) * 3 + j]
                   + l2 * gw[(128 + t) * 3 + j];
        logit[j] = block64_sum(part, red) + gbias[j];
    }
    float mx = fmaxf(logit[0], fmaxf(logit[1], logit[2]));
    float e0 = expf(logit[0] - mx), e1 = expf(logit[1] - mx), e2 = expf(logit[2] - mx);
    float inv = 1.0f / (e0 + e1 + e2);
    float mix = c0 * (e0 * inv) + c1 * (e1 * inv) + c2 * (e2 * inv);
    mix_out[o + t] = mix;

    float m2 = block64_sum(mix, red) * (1.0f / 64.0f);
    float dd = mix - m2;
    float v2 = block64_sum(dd * dd, red) * (1.0f / 64.0f);
    lnmix_out[o + t] = fg[t] * dd * rsqrtf(v2 + 1e-5f) + fb[t];
}

__global__ __launch_bounds__(256) void zero_f32(float* __restrict__ p, int n) {
    int i = blockIdx.x * blockDim.x + threadIdx.x;
    if (i < n) p[i] = 0.0f;
}

// ---------------------------------------------------------------------------
extern "C" void kernel_launch(void* const* d_in, const int* in_sizes, int n_in,
                              void* d_out, int out_size, void* d_ws, size_t ws_size,
                              hipStream_t stream) {
    (void)in_sizes; (void)n_in; (void)out_size; (void)ws_size;
    const float* e      = (const float*)d_in[0];
    const float* u      = (const float*)d_in[1];
    const float* x      = (const float*)d_in[2];
    const float* fe_w1  = (const float*)d_in[3];
    const float* fe_b1  = (const float*)d_in[4];
    const float* fe_w2  = (const float*)d_in[5];
    const float* fe_b2  = (const float*)d_in[6];
    const float* fdel   = (const float*)d_in[7];
    const float* fbias  = (const float*)d_in[8];
    const float* rw     = (const float*)d_in[9];
    const float* rb     = (const float*)d_in[10];
    const float* aw     = (const float*)d_in[11];
    const float* mha_g  = (const float*)d_in[12];
    const float* mha_b  = (const float*)d_in[13];
    const float* wq1    = (const float*)d_in[14];
    const float* bq1    = (const float*)d_in[15];
    const float* wk1    = (const float*)d_in[16];
    const float* bk1    = (const float*)d_in[17];
    const float* wq2    = (const float*)d_in[18];
    const float* bq2    = (const float*)d_in[19];
    const float* wk2    = (const float*)d_in[20];
    const float* bk2    = (const float*)d_in[21];
    const float* wv     = (const float*)d_in[22];
    const float* bv     = (const float*)d_in[23];
    const float* lq1    = (const float*)d_in[24];
    const float* lk1    = (const float*)d_in[25];
    const float* lq2    = (const float*)d_in[26];
    const float* lk2    = (const float*)d_in[27];
    const float* attn_g = (const float*)d_in[28];
    const float* attn_b = (const float*)d_in[29];
    const float* out_w  = (const float*)d_in[30];
    const float* out_b  = (const float*)d_in[31];
    const float* gln_g  = (const float*)d_in[32];
    const float* gln_b  = (const float*)d_in[33];
    const float* gate_w = (const float*)d_in[34];
    const float* gate_b = (const float*)d_in[35];
    const float* ffn_g  = (const float*)d_in[36];
    const float* ffn_b  = (const float*)d_in[37];
    const float* ffn_w1 = (const float*)d_in[38];
    const float* ffn_b1 = (const float*)d_in[39];
    const float* ffn_w2 = (const float*)d_in[40];
    const float* ffn_b2 = (const float*)d_in[41];
    float* outp = (float*)d_out;

    // workspace layout (f32)
    float* ws = (float*)d_ws;
    size_t off = 0;
    auto alloc = [&](size_t n) { float* p = ws + off; off += (n + 15) & ~(size_t)15; return p; };
    float* t     = alloc((size_t)NN * HIDN);
    float* h     = alloc((size_t)NN * CC);
    float* new_e = alloc(NN);
    float* utx   = alloc((size_t)NN * CC);
    float* hfur  = alloc((size_t)NN * CC);
    float* mh    = alloc((size_t)NN * CC);
    float* q1    = alloc((size_t)NN * CC);
    float* k1    = alloc((size_t)NN * CC);
    float* q2    = alloc((size_t)NN * CC);
    float* k2    = alloc((size_t)NN * CC);
    float* v     = alloc((size_t)NN * CC);
    float* k1v   = alloc(CC * CC);     // contiguous with k2v for one zero pass
    float* k2v   = alloc(CC * CC);
    float* xa1   = alloc((size_t)NN * CC);
    float* xa2   = alloc((size_t)NN * CC);
    float* xa    = alloc((size_t)NN * CC);
    float* mhah  = alloc((size_t)NN * CC);
    float* mix   = alloc((size_t)NN * CC);
    float* lnmix = alloc((size_t)NN * CC);
    float* f1    = alloc((size_t)NN * CC);
    float* scal  = alloc(16);

    const dim3 B256(256), B64(64);
    const int MT = NN / 16;            // 625 row tiles

    // --- eig path (independent) ---
    eig_encode<<<(NN + 255) / 256, B256, 0, stream>>>(e, fdel, fbias, rw, rb, aw, new_e, NN);

    // --- feat encoder: t = relu(x@W1+b1); h = t@W2+b2 ---
    gemm_launch(false, false, dim3(MT, 2, 1), stream, x, NFEATN, fe_w1, HIDN,
        nullptr, fe_b1, nullptr, t, HIDN, NN, HIDN, NFEATN, 1, NFEATN, 0);
    gemm_launch(false, false, dim3(MT, 1, 1), stream, t, HIDN, fe_w2, CC,
        nullptr, fe_b2, nullptr, h, CC, NN, CC, HIDN, 0, HIDN, 0);

    // --- spectral propagation: utx = u^T @ h ; h_fur = u @ (new_e .* utx) ---
    gemm_launch(true, false, dim3(MT, 1, 1), stream, u, NN, h, CC,
        nullptr, nullptr, nullptr, utx, CC, NN, CC, NN, 0, NN, 0);
    gemm_launch(false, true, dim3(MT, 1, 1), stream, u, NN, utx, CC,
        new_e, nullptr, nullptr, hfur, CC, NN, CC, NN, 0, NN, 0);

    // --- differential attention ---
    ln_rows64<<<(NN + 7) / 8, B256, 0, stream>>>(h, mha_g, mha_b, mh, NN);
    gemm_launch(false, false, dim3(MT, 1, 1), stream, mh, CC, wq1, CC,
        nullptr, bq1, nullptr, q1, CC, NN, CC, CC, 0, CC, 0);
    gemm_launch(false, false, dim3(MT, 1, 1), stream, mh, CC, wk1, CC,
        nullptr, bk1, nullptr, k1, CC, NN, CC, CC, 0, CC, 0);
    gemm_launch(false, false, dim3(MT, 1, 1), stream, mh, CC, wq2, CC,
        nullptr, bq2, nullptr, q2, CC, NN, CC, CC, 0, CC, 0);
    gemm_launch(false, false, dim3(MT, 1, 1), stream, mh, CC, wk2, CC,
        nullptr, bk2, nullptr, k2, CC, NN, CC, CC, 0, CC, 0);
    gemm_launch(false, false, dim3(MT, 1, 1), stream, mh, CC, wv, CC,
        nullptr, bv, nullptr, v, CC, NN, CC, CC, 0, CC, 0);

    // k1v = k1^T @ v, k2v = k2^T @ v   (split-K over N, atomic f32 accumulate)
    zero_f32<<<(2 * CC * CC + 255) / 256, B256, 0, stream>>>(k1v, 2 * CC * CC);
    gemm_launch(true, false, dim3(4, 1, 40), stream, k1, CC, v, CC,
        nullptr, nullptr, nullptr, k1v, CC, CC, CC, NN, 0, 256, 1);
    gemm_launch(true, false, dim3(4, 1, 40), stream, k2, CC, v, CC,
        nullptr, nullptr, nullptr, k2v, CC, CC, CC, NN, 0, 256, 1);
    lam_kernel<<<1, B64, 0, stream>>>(lq1, lk1, lq2, lk2, scal);

    gemm_launch(false, false, dim3(MT, 1, 1), stream, q1, CC, k1v, CC,
        nullptr, nullptr, nullptr, xa1, CC, NN, CC, CC, 0, CC, 0);
    gemm_launch(false, false, dim3(MT, 1, 1), stream, q2, CC, k2v, CC,
        nullptr, nullptr, nullptr, xa2, CC, NN, CC, CC, 0, CC, 0);
    attn_combine<<<(NN + 7) / 8, B256, 0, stream>>>(xa1, xa2, scal, attn_g, attn_b, xa, NN);
    gemm_launch(false, false, dim3(MT, 1, 1), stream, xa, CC, out_w, CC,
        nullptr, out_b, nullptr, mhah, CC, NN, CC, CC, 0, CC, 0);

    // --- gated mixing + FFN ---
    gate_mix<<<NN, B64, 0, stream>>>(h, mhah, hfur, gln_g, gln_b, gate_w, gate_b,
                                     ffn_g, ffn_b, mix, lnmix, NN);
    gemm_launch(false, false, dim3(MT, 1, 1), stream, lnmix, CC, ffn_w1, CC,
        nullptr, ffn_b1, nullptr, f1, CC, NN, CC, CC, 2, CC, 0);
    gemm_launch(false, false, dim3(MT, 1, 1), stream, f1, CC, ffn_w2, CC,
        nullptr, ffn_b2, mix, outp, CC, NN, CC, CC, 0, CC, 0);
}
